// MAET_21517786153081
// MI455X (gfx1250) — matrix operand order repaired and checked by reading, not compiled, not run on hardware
//
#include <hip/hip_runtime.h>
#include <hip/hip_bf16.h>
#include <math.h>

// ---------------------------------------------------------------------------
// CDNA5 (gfx1250) transformer block, f16 WMMA path (v_wmma_f32_16x16x32_f16),
// register-blocked 2x4 WMMA GEMMs + async global->LDS staged flash attention.
// ---------------------------------------------------------------------------

typedef _Float16 h16;
typedef __attribute__((ext_vector_type(16))) _Float16 v16h;
typedef __attribute__((ext_vector_type(8)))  _Float16 v8h;
typedef __attribute__((ext_vector_type(8)))  float    v8f;

#define WMMA_F32_F16(a, b, c) \
  __builtin_amdgcn_wmma_f32_16x16x32_f16(false, (a), false, (b), (short)0, (c), false, false)

constexpr int Bsz = 8, Nn = 1024, Cc = 768, Hh = 12, Dd = 64, HIDn = 3072, TXT = 512;
constexpr int Mrows = Bsz * Nn;   // 8192 token rows

// ---------------- WMMA fragment loaders (ISA 7.12.2 layouts) ----------------
// A fragment, 16x32 f16, row-major source, leading dim lda (halves).
// lane L: row M = L%16, half h = L/16; elems 0..7 = K=h*8.., elems 8..15 = K=16+h*8..
__device__ __forceinline__ v16h load_frag_a(const h16* A, int lda, int lane) {
  int m = lane & 15, h = lane >> 4;
  v16h r;
  *(v8h*)&r       = *(const v8h*)(A + (size_t)m * lda + h * 8);
  *((v8h*)&r + 1) = *(const v8h*)(A + (size_t)m * lda + 16 + h * 8);
  return r;
}
// B fragment, 32x16 where Bmat[k][n] = W[n][k] (W row-major, ld = ldw halves).
// lane L: col n = L%16, kh = L/16; elems 0..7 = K=kh*8.., elems 8..15 = K=16+kh*8..
__device__ __forceinline__ v16h load_frag_b(const h16* W, int ldw, int lane) {
  int n = lane & 15, kh = lane >> 4;
  v16h r;
  *(v8h*)&r       = *(const v8h*)(W + (size_t)n * ldw + kh * 8);
  *((v8h*)&r + 1) = *(const v8h*)(W + (size_t)n * ldw + 16 + kh * 8);
  return r;
}

// Register-blocked GEMM core: one wave computes a 32x64 tile (2x4 WMMA tiles).
// 12 b128 loads feed 8 WMMAs per K-step (vs 4 loads / 1 WMMA unblocked).
__device__ __forceinline__ void gemm_2x4(const h16* A, int lda, const h16* W, int ldw,
                                         int K, int lane, v8f acc[2][4]) {
  for (int k0 = 0; k0 < K; k0 += 32) {
    v16h a0 = load_frag_a(A + k0, lda, lane);
    v16h a1 = load_frag_a(A + (size_t)16 * lda + k0, lda, lane);
    v16h b0 = load_frag_b(W + k0, ldw, lane);
    v16h b1 = load_frag_b(W + (size_t)16 * ldw + k0, ldw, lane);
    v16h b2 = load_frag_b(W + (size_t)32 * ldw + k0, ldw, lane);
    v16h b3 = load_frag_b(W + (size_t)48 * ldw + k0, ldw, lane);
    acc[0][0] = WMMA_F32_F16(a0, b0, acc[0][0]);
    acc[0][1] = WMMA_F32_F16(a0, b1, acc[0][1]);
    acc[0][2] = WMMA_F32_F16(a0, b2, acc[0][2]);
    acc[0][3] = WMMA_F32_F16(a0, b3, acc[0][3]);
    acc[1][0] = WMMA_F32_F16(a1, b0, acc[1][0]);
    acc[1][1] = WMMA_F32_F16(a1, b1, acc[1][1]);
    acc[1][2] = WMMA_F32_F16(a1, b2, acc[1][2]);
    acc[1][3] = WMMA_F32_F16(a1, b3, acc[1][3]);
  }
}

// ---------------- gfx1250 async global -> LDS (ASYNCcnt-tracked) ----------------
__device__ __forceinline__ void async_g2l_b128(void* lds_dst, const void* gsrc) {
  unsigned ldsoff = (unsigned)(uintptr_t)lds_dst;   // aperture: addr[31:0] = LDS offset
  asm volatile("global_load_async_to_lds_b128 %0, %1, off"
               :: "v"(ldsoff), "v"((unsigned long long)(uintptr_t)gsrc)
               : "memory");
}
__device__ __forceinline__ void wait_async0() {
  asm volatile("s_wait_asynccnt 0x0" ::: "memory");
}

// ---------------- utility kernels ----------------
__global__ void cvt_f16_kernel(const float* __restrict__ s, h16* __restrict__ d, int n) {
  int i = blockIdx.x * 256 + threadIdx.x;
  int stride = gridDim.x * 256;
  for (; i < n; i += stride) d[i] = (h16)s[i];
}

__device__ __forceinline__ float block_sum(float v, float* red) {
  int t = threadIdx.x;
  red[t] = v; __syncthreads();
  for (int s = 128; s > 0; s >>= 1) { if (t < s) red[t] += red[t + s]; __syncthreads(); }
  float r = red[0]; __syncthreads();
  return r;
}

__global__ void ln1_kernel(const float* __restrict__ x, const float* __restrict__ g,
                           const float* __restrict__ bta, h16* __restrict__ out) {
  __shared__ float red[256];
  int row = blockIdx.x, t = threadIdx.x;
  const float* xr = x + (size_t)row * Cc;
  float a0 = xr[t], a1 = xr[t + 256], a2 = xr[t + 512];
  float mu = block_sum(a0 + a1 + a2, red) * (1.f / Cc);
  float d0 = a0 - mu, d1 = a1 - mu, d2 = a2 - mu;
  float var = block_sum(d0 * d0 + d1 * d1 + d2 * d2, red) * (1.f / Cc);
  float rs = rsqrtf(var + 1e-5f);
  h16* orow = out + (size_t)row * Cc;
  orow[t]       = (h16)(d0 * rs * g[t] + bta[t]);
  orow[t + 256] = (h16)(d1 * rs * g[t + 256] + bta[t + 256]);
  orow[t + 512] = (h16)(d2 * rs * g[t + 512] + bta[t + 512]);
}

__global__ void ln2_kernel(const float* __restrict__ x,
                           const float* gt, const float* bt,
                           const float* gi, const float* bi, h16* __restrict__ out) {
  __shared__ float red[256];
  int row = blockIdx.x, t = threadIdx.x;
  bool text = (row & (Nn - 1)) < TXT;
  const float* g   = text ? gt : gi;
  const float* bta = text ? bt : bi;
  const float* xr = x + (size_t)row * Cc;
  float a0 = xr[t], a1 = xr[t + 256], a2 = xr[t + 512];
  float mu = block_sum(a0 + a1 + a2, red) * (1.f / Cc);
  float d0 = a0 - mu, d1 = a1 - mu, d2 = a2 - mu;
  float var = block_sum(d0 * d0 + d1 * d1 + d2 * d2, red) * (1.f / Cc);
  float rs = rsqrtf(var + 1e-5f);
  h16* orow = out + (size_t)row * Cc;
  orow[t]       = (h16)(d0 * rs * g[t] + bta[t]);
  orow[t + 256] = (h16)(d1 * rs * g[t + 256] + bta[t + 256]);
  orow[t + 512] = (h16)(d2 * rs * g[t + 512] + bta[t + 512]);
}

// ---------------- GEMM kernels (1 wave = 32x64 output tile) ----------------
__global__ void gemm_qkv_kernel(const h16* __restrict__ A, const h16* __restrict__ W,
                                const float* __restrict__ qb, const float* __restrict__ vb,
                                h16* __restrict__ q16, h16* __restrict__ k16,
                                h16* __restrict__ vT16) {
  int lane = threadIdx.x & 31, wv = threadIdx.x >> 5;
  int tile = blockIdx.x * 8 + wv;
  const int NT = (3 * Cc) / 64;                       // 36
  int mt = tile / NT, nt = tile % NT;
  int row0 = mt * 32, col0 = nt * 64;
  v8f acc[2][4] = {};
  gemm_2x4(A + (size_t)row0 * Cc, Cc, W + (size_t)col0 * Cc, Cc, Cc, lane, acc);
  int nl = lane & 15, hf = lane >> 4;
  int part = col0 / Cc;                               // uniform per 64-col block
  int w0 = col0 % Cc, head = w0 >> 6;
  for (int ni = 0; ni < 4; ++ni) {
    int w = w0 + ni * 16 + nl;
    int d = ni * 16 + nl;
    float bias = (part == 0) ? qb[w] : (part == 2 ? vb[w] : 0.f);
    for (int mi = 0; mi < 2; ++mi)
      for (int r = 0; r < 8; ++r) {
        int gm = row0 + mi * 16 + hf * 8 + r;
        int b = gm >> 10, tok = gm & (Nn - 1);
        float val = acc[mi][ni][r] + bias;
        size_t idx = (((size_t)(b * Hh + head)) * Nn + tok) * Dd + d;
        if (part == 0)      q16[idx] = (h16)(val * 0.125f);      // * D^-0.5
        else if (part == 1) k16[idx] = (h16)val;
        else vT16[(((size_t)(b * Hh + head)) * Dd + d) * Nn + tok] = (h16)val;
      }
  }
}

// Attention: 1 block = (b, h, 128 query rows); 8 waves, one 16-row tile each.
// K/V chunks staged cooperatively in LDS via async global->LDS.
__global__ void attn_kernel(const h16* __restrict__ q16, const h16* __restrict__ k16,
                            const h16* __restrict__ vT16,
                            const float* __restrict__ rel_bias,
                            const unsigned char* __restrict__ mask,  // jnp bool_ = 1 byte
                            h16* __restrict__ O16) {
  __shared__ __align__(16) h16 Ks[32 * 64];           // [key][d], 4KB
  __shared__ __align__(16) h16 Vs[64 * 32];           // [d][key], 4KB
  __shared__ __align__(16) h16 plds[8][16 * 32];      // per-wave P staging, 8KB
  int tid = threadIdx.x;
  int lane = tid & 31, wv = tid >> 5;
  int blk = blockIdx.x;                               // 0..767
  int b = blk / (Hh * 8);
  int rem = blk % (Hh * 8);
  int h = rem / 8;
  int q0 = (rem % 8) * 128 + wv * 16;
  int bh = b * Hh + h;
  const h16* qbase = q16 + ((size_t)bh * Nn + q0) * Dd;
  v16h aq0 = load_frag_a(qbase, Dd, lane);            // d = 0..31
  v16h aq1 = load_frag_a(qbase + 32, Dd, lane);       // d = 32..63
  int nl = lane & 15, hf = lane >> 4;
  const float NEG = -__builtin_huge_valf();
  float m[8], l[8];
  v8f o0 = {}, o1 = {}, o2 = {}, o3 = {};
  for (int r = 0; r < 8; ++r) { m[r] = NEG; l[r] = 0.f; }
  const float* bb = rel_bias + ((size_t)h * Nn + q0) * Nn;
  const unsigned char* mk = mask + (size_t)b * Nn;
  h16* pw = &plds[wv][0];
  // per-thread staging slots (16B each, 256 threads cover 4KB per buffer)
  h16* ksDst = &Ks[tid * 8];
  h16* vsDst = &Vs[(tid >> 2) * 32 + (tid & 3) * 8];

  for (int ck = 0; ck < Nn / 32; ++ck) {
    int m0 = ck * 32;
    // ---- async stage K (32x64, row-major) and V^T (64x32, keys contiguous) ----
    async_g2l_b128(ksDst, k16 + ((size_t)bh * Nn + m0) * Dd + tid * 8);
    async_g2l_b128(vsDst, vT16 + ((size_t)(bh * Dd) + (tid >> 2)) * Nn + m0 + (tid & 3) * 8);
    wait_async0();
    __syncthreads();

    // ---- S = q @ K^T for 2 key sub-tiles ----
    v8f s0 = {}, s1 = {};
    s0 = WMMA_F32_F16(aq0, load_frag_b(Ks, Dd, lane), s0);              // keys m0..+15
    s0 = WMMA_F32_F16(aq1, load_frag_b(Ks + 32, Dd, lane), s0);
    s1 = WMMA_F32_F16(aq0, load_frag_b(Ks + 16 * Dd, Dd, lane), s1);    // keys +16..+31
    s1 = WMMA_F32_F16(aq1, load_frag_b(Ks + 16 * Dd + 32, Dd, lane), s1);

    int key0 = m0 + nl, key1 = m0 + 16 + nl;
    bool ok0 = mk[key0] != 0, ok1 = mk[key1] != 0;
    for (int r = 0; r < 8; ++r) {
      int row = hf * 8 + r;
      float t0 = ok0 ? s0[r] + bb[(size_t)row * Nn + key0] : NEG;
      float t1 = ok1 ? s1[r] + bb[(size_t)row * Nn + key1] : NEG;
      float cm = fmaxf(t0, t1);
      cm = fmaxf(cm, __shfl_xor(cm, 1));
      cm = fmaxf(cm, __shfl_xor(cm, 2));
      cm = fmaxf(cm, __shfl_xor(cm, 4));
      cm = fmaxf(cm, __shfl_xor(cm, 8));              // row max over 16 lanes
      float mn = fmaxf(m[r], cm);
      float p0 = (mn == NEG) ? 0.f : __expf(t0 - mn);
      float p1 = (mn == NEG) ? 0.f : __expf(t1 - mn);
      float alpha = (mn == NEG) ? 1.f : __expf(m[r] - mn);
      float rs = p0 + p1;
      rs += __shfl_xor(rs, 1); rs += __shfl_xor(rs, 2);
      rs += __shfl_xor(rs, 4); rs += __shfl_xor(rs, 8);
      l[r] = l[r] * alpha + rs;
      m[r] = mn;
      o0[r] *= alpha; o1[r] *= alpha; o2[r] *= alpha; o3[r] *= alpha;
      pw[row * 32 + nl]      = (h16)p0;               // C/D layout -> LDS row-major
      pw[row * 32 + 16 + nl] = (h16)p1;
    }
    __builtin_amdgcn_wave_barrier();
    v16h pa = load_frag_a(pw, 32, lane);              // re-read as A fragment (K=32 keys)
    __builtin_amdgcn_wave_barrier();
    o0 = WMMA_F32_F16(pa, load_frag_b(Vs + 0 * 16 * 32, 32, lane), o0);
    o1 = WMMA_F32_F16(pa, load_frag_b(Vs + 1 * 16 * 32, 32, lane), o1);
    o2 = WMMA_F32_F16(pa, load_frag_b(Vs + 2 * 16 * 32, 32, lane), o2);
    o3 = WMMA_F32_F16(pa, load_frag_b(Vs + 3 * 16 * 32, 32, lane), o3);
    __syncthreads();                                  // Ks/Vs reusable next chunk
  }
  for (int r = 0; r < 8; ++r) {
    float inv = (l[r] > 0.f) ? 1.f / l[r] : 0.f;
    int gm = b * Nn + q0 + hf * 8 + r;
    h16* orow = O16 + (size_t)gm * Cc + h * Dd;
    orow[0 * 16 + nl] = (h16)(o0[r] * inv);
    orow[1 * 16 + nl] = (h16)(o1[r] * inv);
    orow[2 * 16 + nl] = (h16)(o2[r] * inv);
    orow[3 * 16 + nl] = (h16)(o3[r] * inv);
  }
}

__global__ void gemm_proj_kernel(const h16* __restrict__ A, const h16* __restrict__ W,
                                 const float* __restrict__ bp, const float* __restrict__ x,
                                 const float* __restrict__ gamma1, float* __restrict__ x1) {
  int lane = threadIdx.x & 31, wv = threadIdx.x >> 5;
  int tile = blockIdx.x * 8 + wv;
  const int NT = Cc / 64;                              // 12
  int mt = tile / NT, nt = tile % NT;
  int row0 = mt * 32, col0 = nt * 64;
  v8f acc[2][4] = {};
  gemm_2x4(A + (size_t)row0 * Cc, Cc, W + (size_t)col0 * Cc, Cc, Cc, lane, acc);
  int nl = lane & 15, hf = lane >> 4;
  for (int ni = 0; ni < 4; ++ni) {
    int c = col0 + ni * 16 + nl;
    for (int mi = 0; mi < 2; ++mi)
      for (int r = 0; r < 8; ++r) {
        int gm = row0 + mi * 16 + hf * 8 + r;
        size_t o = (size_t)gm * Cc + c;
        x1[o] = x[o] + gamma1[c] * (acc[mi][ni][r] + bp[c]);
      }
  }
}

__global__ void gemm_mlp1_kernel(const h16* __restrict__ A,
                                 const h16* __restrict__ W1t, const h16* __restrict__ W1i,
                                 const float* __restrict__ b1t, const float* __restrict__ b1i,
                                 h16* __restrict__ g) {
  int lane = threadIdx.x & 31, wv = threadIdx.x >> 5;
  int tile = blockIdx.x * 8 + wv;
  const int NT = HIDn / 64;                            // 48
  int mt = tile / NT, nt = tile % NT;
  int row0 = mt * 32, col0 = nt * 64;
  bool text = (row0 & (Nn - 1)) < TXT;                 // uniform per 32-row block
  const h16* W = text ? W1t : W1i;
  const float* b1 = text ? b1t : b1i;
  v8f acc[2][4] = {};
  gemm_2x4(A + (size_t)row0 * Cc, Cc, W + (size_t)col0 * Cc, Cc, Cc, lane, acc);
  int nl = lane & 15, hf = lane >> 4;
  for (int ni = 0; ni < 4; ++ni) {
    int c = col0 + ni * 16 + nl;
    for (int mi = 0; mi < 2; ++mi)
      for (int r = 0; r < 8; ++r) {
        int gm = row0 + mi * 16 + hf * 8 + r;
        float v = acc[mi][ni][r] + b1[c];
        float ge = 0.5f * v * (1.f + erff(v * 0.70710678118654752f));   // exact GELU
        g[(size_t)gm * HIDn + c] = (h16)ge;
      }
  }
}

__global__ void gemm_mlp2_kernel(const h16* __restrict__ A,
                                 const h16* __restrict__ W2t, const h16* __restrict__ W2i,
                                 const float* __restrict__ b2t, const float* __restrict__ b2i,
                                 const float* __restrict__ x1, const float* __restrict__ gamma2,
                                 float* __restrict__ out) {
  int lane = threadIdx.x & 31, wv = threadIdx.x >> 5;
  int tile = blockIdx.x * 8 + wv;
  const int NT = Cc / 64;                              // 12
  int mt = tile / NT, nt = tile % NT;
  int row0 = mt * 32, col0 = nt * 64;
  bool text = (row0 & (Nn - 1)) < TXT;
  const h16* W = text ? W2t : W2i;
  const float* b2 = text ? b2t : b2i;
  v8f acc[2][4] = {};
  gemm_2x4(A + (size_t)row0 * HIDn, HIDn, W + (size_t)col0 * HIDn, HIDn, HIDn, lane, acc);
  int nl = lane & 15, hf = lane >> 4;
  for (int ni = 0; ni < 4; ++ni) {
    int c = col0 + ni * 16 + nl;
    for (int mi = 0; mi < 2; ++mi)
      for (int r = 0; r < 8; ++r) {
        int gm = row0 + mi * 16 + hf * 8 + r;
        size_t o = (size_t)gm * Cc + c;
        out[o] = x1[o] + gamma2[c] * (acc[mi][ni][r] + b2[c]);
      }
  }
}

// ---------------------------------------------------------------------------
extern "C" void kernel_launch(void* const* d_in, const int* in_sizes, int n_in,
                              void* d_out, int out_size, void* d_ws, size_t ws_size,
                              hipStream_t stream) {
  const float* x       = (const float*)d_in[0];
  const unsigned char* mask = (const unsigned char*)d_in[1];   // jnp bool_ (1 byte)
  const float* rel_bias = (const float*)d_in[2];
  const float* w_qkv   = (const float*)d_in[3];
  const float* q_bias  = (const float*)d_in[4];
  const float* v_bias  = (const float*)d_in[5];
  const float* w_proj  = (const float*)d_in[6];
  const float* b_proj  = (const float*)d_in[7];
  const float* ln1_g   = (const float*)d_in[8];
  const float* ln1_b   = (const float*)d_in[9];
  const float* ln2t_g  = (const float*)d_in[10];
  const float* ln2t_b  = (const float*)d_in[11];
  const float* ln2i_g  = (const float*)d_in[12];
  const float* ln2i_b  = (const float*)d_in[13];
  const float* w1t     = (const float*)d_in[14];
  const float* b1t     = (const float*)d_in[15];
  const float* w2t     = (const float*)d_in[16];
  const float* b2t     = (const float*)d_in[17];
  const float* w1i     = (const float*)d_in[18];
  const float* b1i     = (const float*)d_in[19];
  const float* w2i     = (const float*)d_in[20];
  const float* b2i     = (const float*)d_in[21];
  const float* gamma1  = (const float*)d_in[22];
  const float* gamma2  = (const float*)d_in[23];
  float* out = (float*)d_out;

  // workspace carve-up (256B aligned); hF/q/k/vT block is reused for MLP hidden g.
  char* ws = (char*)d_ws;
  auto take = [&](size_t bytes) { char* p = ws; ws += (bytes + 255) & ~(size_t)255; return p; };
  const size_t tokC2 = (size_t)Mrows * Cc * 2;                 // 12.58 MB
  h16* hF   = (h16*)take(tokC2);                               // ln1 out; later reused as O
  h16* q16  = (h16*)take(tokC2);
  h16* k16  = (h16*)take(tokC2);
  h16* vT16 = (h16*)take(tokC2);
  h16* g16  = hF;                                              // 4*tokC2 contiguous = g [8192x3072]
  h16* O16  = hF;                                              // O reuses hF after QKV GEMM
  h16* wqkv16  = (h16*)take((size_t)3 * Cc * Cc * 2);
  h16* wproj16 = (h16*)take((size_t)Cc * Cc * 2);
  h16* w1t16 = (h16*)take((size_t)HIDn * Cc * 2);
  h16* w2t16 = (h16*)take((size_t)HIDn * Cc * 2);
  h16* w1i16 = (h16*)take((size_t)HIDn * Cc * 2);
  h16* w2i16 = (h16*)take((size_t)HIDn * Cc * 2);
  float* x1 = (float*)take((size_t)Mrows * Cc * 4);
  h16* h2   = (h16*)take(tokC2);

  // 1) weights -> f16
  cvt_f16_kernel<<<2048, 256, 0, stream>>>(w_qkv,  wqkv16,  3 * Cc * Cc);
  cvt_f16_kernel<<<2048, 256, 0, stream>>>(w_proj, wproj16, Cc * Cc);
  cvt_f16_kernel<<<2048, 256, 0, stream>>>(w1t, w1t16, HIDn * Cc);
  cvt_f16_kernel<<<2048, 256, 0, stream>>>(w2t, w2t16, HIDn * Cc);
  cvt_f16_kernel<<<2048, 256, 0, stream>>>(w1i, w1i16, HIDn * Cc);
  cvt_f16_kernel<<<2048, 256, 0, stream>>>(w2i, w2i16, HIDn * Cc);

  // 2) LN1
  ln1_kernel<<<Mrows, 256, 0, stream>>>(x, ln1_g, ln1_b, hF);
  // 3) QKV GEMM: 256 row-blocks * 36 col-blocks / 8 waves
  gemm_qkv_kernel<<<1152, 256, 0, stream>>>(hF, wqkv16, q_bias, v_bias, q16, k16, vT16);
  // 4) attention: B*H*(N/128) blocks
  attn_kernel<<<768, 256, 0, stream>>>(q16, k16, vT16, rel_bias, mask, O16);
  // 5) proj + gamma1 residual: 256*12/8
  gemm_proj_kernel<<<384, 256, 0, stream>>>(O16, wproj16, b_proj, x, gamma1, x1);
  // 6) LN2 (text/image params)
  ln2_kernel<<<Mrows, 256, 0, stream>>>(x1, ln2t_g, ln2t_b, ln2i_g, ln2i_b, h2);
  // 7) MLP fc1 + GELU: 256*48/8
  gemm_mlp1_kernel<<<1536, 256, 0, stream>>>(h2, w1t16, w1i16, b1t, b1i, g16);
  // 8) MLP fc2 + gamma2 residual -> final output: 256*12/8
  gemm_mlp2_kernel<<<384, 256, 0, stream>>>(g16, w2t16, w2i16, b2t, b2i, x1, gamma2, out);
}
